// drugpt_36558761623925
// MI455X (gfx1250) — matrix operand order repaired and checked by reading, not compiled, not run on hardware
//
#include <hip/hip_runtime.h>
#include <hip/hip_bf16.h>

typedef __attribute__((ext_vector_type(16))) _Float16 v16h;
typedef __attribute__((ext_vector_type(8)))  _Float16 v8h;
typedef __attribute__((ext_vector_type(8)))  float    v8f;

// ---------------- problem constants ----------------
// B=32, G=2048, D=2048, E=64, H=6, NL=16, NM=4, NGPL=32, DH=[100,50,6], NF=6
#define XW 4097            // G + D + 1
#define EPSV 1e-5f

// ---------------- workspace layout (bytes) ----------------
static constexpr size_t OFF_RBFP    = 0;                       // 64*4*32*16 halves  = 256 KB
static constexpr size_t OFF_LEAFP   = OFF_RBFP   + 262144;     // 64*32*32*16 halves = 2 MB
static constexpr size_t OFF_GPERM   = OFF_LEAFP  + 2097152;    // 32*64*2048 f16     = 8 MB
static constexpr size_t OFF_CILEAF  = OFF_GPERM  + 8388608;    // 32*16*64*32 f32    = 4 MB
static constexpr size_t OFF_DRUG    = OFF_CILEAF + 4194304;    // 32*6 f32
static constexpr size_t OFF_LEAFOUT = OFF_DRUG   + 1024;       // 32*16*6*64 f32     = 768 KB
static constexpr size_t OFF_LEAFAUX = OFF_LEAFOUT+ 786432;     // 32*16 f32
static constexpr size_t OFF_MIDOUT  = OFF_LEAFAUX+ 2048;       // 32*4*6*64 f32
static constexpr size_t OFF_MIDAUX  = OFF_MIDOUT + 196608;     // 32*4 f32
static constexpr size_t OFF_RMEAN   = OFF_MIDAUX + 1024;       // 32*64 f32
static constexpr size_t OFF_RAUX    = OFF_RMEAN  + 8192;       // 32 f32

// =====================================================================
// K0: pack rbf_W (64,2048) and leaf_dg_W (16*32,2048) into f16 WMMA
// B-matrix tile layout: [ktile][ntile][lane][16 halves]
//   lane<16 : column N=lane,    element i -> K = ktile*32 + i (i<8), ktile*32 + 8 + i (i>=8)
//   lane>=16: column N=lane-16, element i -> K = ktile*32 + 8 + i (i<8), ktile*32 + 16 + i (i>=8)
// =====================================================================
__global__ __launch_bounds__(256) void pack_weights_kernel(
    const float* __restrict__ rbf_W, const float* __restrict__ leaf_dg_W,
    _Float16* __restrict__ rbfW_pack, _Float16* __restrict__ leafW_pack)
{
  const int nthr = gridDim.x * 256;
  const int tid0 = blockIdx.x * 256 + threadIdx.x;
  for (int idx = tid0; idx < 64*4*32*16; idx += nthr) {
    int i = idx & 15, lane = (idx >> 4) & 31, nt = (idx >> 9) & 3, ki = idx >> 11;
    int k = ki * 32 + ((lane & 16) ? 8 : 0) + (i & 7) + ((i & 8) ? 16 : 0);
    int e = nt * 16 + (lane & 15);
    rbfW_pack[idx] = (_Float16)rbf_W[e * 2048 + k];
  }
  for (int idx = tid0; idx < 64*32*32*16; idx += nthr) {
    int i = idx & 15, lane = (idx >> 4) & 31, nt = (idx >> 9) & 31, ki = idx >> 14;
    int k = ki * 32 + ((lane & 16) ? 8 : 0) + (i & 7) + ((i & 8) ? 16 : 0);
    int col = nt * 16 + (lane & 15);                 // col = t*32 + n
    leafW_pack[idx] = (_Float16)leaf_dg_W[(size_t)col * 2048 + k];
  }
}

// =====================================================================
// K1: drug MLP (3 layers, tanh + batchnorm over batch), single block
// =====================================================================
__global__ __launch_bounds__(256) void drug_mlp_kernel(
    const float* __restrict__ x,
    const float* dW0, const float* db0, const float* dg0, const float* dbt0,
    const float* dW1, const float* db1, const float* dg1, const float* dbt1,
    const float* dW2, const float* db2, const float* dg2, const float* dbt2,
    float* __restrict__ drug_out)
{
  __shared__ float h0[32*100];
  __shared__ float h1[32*50];
  __shared__ float h2[32*6];
  const int tid = threadIdx.x;
  for (int o = tid; o < 32*100; o += 256) {
    int b = o / 100, j = o % 100;
    const float* xb = x + (size_t)b * XW + 2048;
    const float* w  = dW0 + (size_t)j * 2048;
    float s = db0[j];
    for (int k = 0; k < 2048; ++k) s += xb[k] * w[k];
    h0[o] = tanhf(s);
  }
  __syncthreads();
  for (int j = tid; j < 100; j += 256) {
    float m = 0.f; for (int b = 0; b < 32; ++b) m += h0[b*100+j]; m *= (1.f/32.f);
    float v = 0.f; for (int b = 0; b < 32; ++b) { float d = h0[b*100+j]-m; v += d*d; } v *= (1.f/32.f);
    float rs = rsqrtf(v + EPSV);
    for (int b = 0; b < 32; ++b) h0[b*100+j] = (h0[b*100+j]-m)*rs*dg0[j] + dbt0[j];
  }
  __syncthreads();
  for (int o = tid; o < 32*50; o += 256) {
    int b = o / 50, j = o % 50;
    const float* w = dW1 + j*100;
    float s = db1[j];
    for (int k = 0; k < 100; ++k) s += h0[b*100+k] * w[k];
    h1[o] = tanhf(s);
  }
  __syncthreads();
  for (int j = tid; j < 50; j += 256) {
    float m = 0.f; for (int b = 0; b < 32; ++b) m += h1[b*50+j]; m *= (1.f/32.f);
    float v = 0.f; for (int b = 0; b < 32; ++b) { float d = h1[b*50+j]-m; v += d*d; } v *= (1.f/32.f);
    float rs = rsqrtf(v + EPSV);
    for (int b = 0; b < 32; ++b) h1[b*50+j] = (h1[b*50+j]-m)*rs*dg1[j] + dbt1[j];
  }
  __syncthreads();
  for (int o = tid; o < 32*6; o += 256) {
    int b = o / 6, j = o % 6;
    float s = db2[j];
    for (int k = 0; k < 50; ++k) s += h1[b*50+k] * dW2[j*50+k];
    h2[o] = tanhf(s);
  }
  __syncthreads();
  for (int j = tid; j < 6; j += 256) {
    float m = 0.f; for (int b = 0; b < 32; ++b) m += h2[b*6+j]; m *= (1.f/32.f);
    float v = 0.f; for (int b = 0; b < 32; ++b) { float d = h2[b*6+j]-m; v += d*d; } v *= (1.f/32.f);
    float rs = rsqrtf(v + EPSV);
    for (int b = 0; b < 32; ++b) drug_out[b*6+j] = (h2[b*6+j]-m)*rs*dg2[j] + dbt2[j];
  }
}

// =====================================================================
// K2: fused RBF kernel-matrix x rbf_W^T GEMM via WMMA f16.
// grid (16, B); block 256 = 8 waves; wave -> 16 output rows, 4 N-tiles.
// B-tile loads issued BEFORE the exp/cvt chain so the ~56 VALU/trans ops
// hide load latency; one consolidated wait before the 4-WMMA cluster.
// Epilogue adds rbf_b + emb LUT and writes gperm (B,E,G) as f16.
// =====================================================================
__global__ __launch_bounds__(256) void rbf_wmma_kernel(
    const float* __restrict__ x, const int* __restrict__ table,
    const float* __restrict__ emb, const float* __restrict__ rbf_b,
    const _Float16* __restrict__ rbfW_pack, _Float16* __restrict__ gperm_h)
{
  __shared__ float sgene[2048];
  const int b   = blockIdx.y;
  const int tid = threadIdx.x;
  const float* xb = x + (size_t)b * XW;
  for (int i = tid; i < 2048; i += 256) sgene[i] = xb[i];
  __syncthreads();

  const int wave  = tid >> 5;            // wave32
  const int lane  = tid & 31;
  const int gbase = blockIdx.x * 128 + wave * 16;
  const int khalf = (lane & 16) ? 8 : 0; // A layout: lanes 16-31 hold K+8
  const float a_g = sgene[gbase + (lane & 15)];
  const float cexp = -0.14426950408889634f;  // -GAMMA * log2(e)

  v8f acc[4] = {};
  const v16h* bbase = (const v16h*)rbfW_pack;
  for (int ki = 0; ki < 64; ++ki) {
    // ---- issue B-tile loads first (8 x global_load_b128) ----
    const v16h* bp = bbase + (size_t)(ki * 4) * 32 + lane;
    v16h bv0 = bp[0];
    v16h bv1 = bp[32];
    v16h bv2 = bp[64];
    v16h bv3 = bp[96];
    __builtin_prefetch(bp + 128, 0, 1);   // next ktile (global_prefetch_b8)

    // ---- A fragment: 16 exps per lane, overlapping the loads ----
    const int kb = ki * 32;
    const float4* p0 = (const float4*)(sgene + kb + khalf);
    const float4* p1 = (const float4*)(sgene + kb + 16 + khalf);
    float4 c0 = p0[0], c1 = p0[1], c2 = p1[0], c3 = p1[1];
    float kv[16];
    kv[0]=c0.x; kv[1]=c0.y; kv[2]=c0.z; kv[3]=c0.w;
    kv[4]=c1.x; kv[5]=c1.y; kv[6]=c1.z; kv[7]=c1.w;
    kv[8]=c2.x; kv[9]=c2.y; kv[10]=c2.z; kv[11]=c2.w;
    kv[12]=c3.x; kv[13]=c3.y; kv[14]=c3.z; kv[15]=c3.w;
    v16h av;
#pragma unroll
    for (int i = 0; i < 16; ++i) {
      float dd = a_g - kv[i];
      av[i] = (_Float16)__builtin_amdgcn_exp2f(cexp * dd * dd);
    }

    // ---- 4 back-to-back WMMAs ----
    acc[0] = __builtin_amdgcn_wmma_f32_16x16x32_f16(false, av, false, bv0, (short)0, acc[0], false, false);
    acc[1] = __builtin_amdgcn_wmma_f32_16x16x32_f16(false, av, false, bv1, (short)0, acc[1], false, false);
    acc[2] = __builtin_amdgcn_wmma_f32_16x16x32_f16(false, av, false, bv2, (short)0, acc[2], false, false);
    acc[3] = __builtin_amdgcn_wmma_f32_16x16x32_f16(false, av, false, bv3, (short)0, acc[3], false, false);
  }

  const int didx = (int)xb[4096];
  const int e_lo = lane & 15;
  const int mo   = (lane & 16) ? 8 : 0;  // C layout: lanes 16-31 hold M+8
#pragma unroll
  for (int nt = 0; nt < 4; ++nt) {
    const int e = nt * 16 + e_lo;
    const float rbv = rbf_b[e];
#pragma unroll
    for (int r = 0; r < 8; ++r) {
      const int g  = gbase + mo + r;
      const int gi = table[didx * 2048 + g];
      float val = acc[nt][r] + rbv + emb[(size_t)gi * 64 + e];
      gperm_h[((size_t)b * 64 + e) * 2048 + g] = (_Float16)val;
    }
  }
}

// =====================================================================
// K3: ci_leaf = gperm (2048 rows x 2048 K) x leaf_dg_W^T (2048 x 512) via WMMA
// grid (16, 8); wave -> 16 rows x 4 N-tiles. Software-pipelined: next
// iteration's A fragment + 4 B tiles are loaded while current WMMAs run.
// =====================================================================
__global__ __launch_bounds__(256) void leaf_wmma_kernel(
    const _Float16* __restrict__ gperm_h, const _Float16* __restrict__ leafW_pack,
    float* __restrict__ ci_leaf)
{
  const int tid  = threadIdx.x;
  const int wave = tid >> 5, lane = tid & 31;
  const int rowbase = blockIdx.x * 128 + wave * 16;
  const int ntbase  = blockIdx.y * 4;
  const int khalf   = (lane & 16) ? 8 : 0;
  const _Float16* arow = gperm_h + (size_t)(rowbase + (lane & 15)) * 2048;
  const v16h* bbase = (const v16h*)leafW_pack;

  // prologue: load iteration 0 fragments
  v8h lo = *(const v8h*)(arow + khalf);
  v8h hi = *(const v8h*)(arow + 16 + khalf);
  const v16h* bp0 = bbase + (size_t)ntbase * 32 + lane;
  v16h b0 = bp0[0], b1 = bp0[32], b2 = bp0[64], b3 = bp0[96];

  v8f acc[4] = {};
  for (int ki = 0; ki < 64; ++ki) {
    v16h av = __builtin_shufflevector(lo, hi, 0,1,2,3,4,5,6,7,8,9,10,11,12,13,14,15);
    v16h c0 = b0, c1 = b1, c2 = b2, c3 = b3;

    // issue next iteration's loads before consuming current tiles
    const int kn = (ki < 63) ? ki + 1 : 63;
    const int kb = kn * 32;
    lo = *(const v8h*)(arow + kb + khalf);
    hi = *(const v8h*)(arow + kb + 16 + khalf);
    const v16h* bp = bbase + ((size_t)kn * 32 + ntbase) * 32 + lane;
    b0 = bp[0]; b1 = bp[32]; b2 = bp[64]; b3 = bp[96];

    acc[0] = __builtin_amdgcn_wmma_f32_16x16x32_f16(false, av, false, c0, (short)0, acc[0], false, false);
    acc[1] = __builtin_amdgcn_wmma_f32_16x16x32_f16(false, av, false, c1, (short)0, acc[1], false, false);
    acc[2] = __builtin_amdgcn_wmma_f32_16x16x32_f16(false, av, false, c2, (short)0, acc[2], false, false);
    acc[3] = __builtin_amdgcn_wmma_f32_16x16x32_f16(false, av, false, c3, (short)0, acc[3], false, false);
  }

  const int mo = (lane & 16) ? 8 : 0;
#pragma unroll
  for (int nt = 0; nt < 4; ++nt) {
    const int col = (ntbase + nt) * 16 + (lane & 15);
    const int t = col >> 5, n = col & 31;
#pragma unroll
    for (int r = 0; r < 8; ++r) {
      const int row = rowbase + mo + r;
      const int bb = row >> 6, e = row & 63;
      ci_leaf[((size_t)(bb * 16 + t) * 64 + e) * 32 + n] = acc[nt][r];
    }
  }
}

// =====================================================================
// K4: leaf term block. grid = 16 (one block per t).
// =====================================================================
__global__ __launch_bounds__(256) void leaf_block_kernel(
    const float* __restrict__ ci, const float* __restrict__ dgb,
    const float* __restrict__ W, const float* __restrict__ bb,
    const float* __restrict__ gg, const float* __restrict__ bt,
    const float* __restrict__ gW, const float* __restrict__ gb,
    const float* __restrict__ a1W, const float* __restrict__ a1b,
    const float* __restrict__ a2w, const float* __restrict__ a2b,
    const float* __restrict__ drug_out,
    float* __restrict__ leaf_out, float* __restrict__ leaf_aux)
{
  const int t = blockIdx.x;
  const int tid = threadIdx.x;
  __shared__ float zs[6][2048];
  __shared__ float red[256], red2[256];
  __shared__ float mh[6], rsh[6];
  __shared__ float bnm[2048];
  __shared__ float gate[32];

  for (int idx = tid; idx < 6*2048; idx += 256) {
    int h = idx >> 11, be = idx & 2047, b = be >> 6, e = be & 63;
    const float* cp = ci + ((size_t)(b*16 + t) * 64 + e) * 32;
    const float* wp = W  + (t*6 + h) * 32;
    const float* dp = dgb + t*32;
    float s = bb[t*6 + h];
#pragma unroll
    for (int n = 0; n < 32; ++n) s += (cp[n] + dp[n]) * wp[n];
    zs[h][be] = tanhf(s);
  }
  __syncthreads();
  for (int h = 0; h < 6; ++h) {
    float s = 0.f, s2 = 0.f;
    for (int i = tid; i < 2048; i += 256) { float v = zs[h][i]; s += v; s2 += v*v; }
    red[tid] = s; red2[tid] = s2; __syncthreads();
    for (int st = 128; st > 0; st >>= 1) {
      if (tid < st) { red[tid] += red[tid+st]; red2[tid] += red2[tid+st]; }
      __syncthreads();
    }
    if (tid == 0) {
      float m = red[0] * (1.f/2048.f);
      float var = red2[0] * (1.f/2048.f) - m*m;
      mh[h] = m; rsh[h] = rsqrtf(var + EPSV);
    }
    __syncthreads();
  }
  for (int be = tid; be < 2048; be += 256) {
    float s = 0.f;
#pragma unroll
    for (int h = 0; h < 6; ++h)
      s += (zs[h][be] - mh[h]) * rsh[h] * gg[t*6+h] + bt[t*6+h];
    bnm[be] = s * (1.f/6.f);
  }
  __syncthreads();
  if (tid < 32) {
    int b = tid;
    float s = gb[t];
    for (int e = 0; e < 64; ++e) s += bnm[b*64+e] * gW[t*70 + e];
    for (int j = 0; j < 6; ++j)  s += drug_out[b*6+j] * gW[t*70 + 64 + j];
    gate[b] = 1.f / (1.f + expf(-s));
  }
  __syncthreads();
  for (int idx = tid; idx < 6*2048; idx += 256) {
    int h = idx >> 11, be = idx & 2047, b = be >> 6, e = be & 63;
    float bn = (zs[h][be] - mh[h]) * rsh[h] * gg[t*6+h] + bt[t*6+h];
    leaf_out[((size_t)(b*16 + t) * 6 + h) * 64 + e] = gate[b] * bn;
  }
  if (tid < 32) {
    int b = tid;
    float s = a1b[t];
    for (int e = 0; e < 64; ++e) s += gate[b] * bnm[b*64+e] * a1W[t*64+e];
    leaf_aux[b*16 + t] = tanhf(s) * a2w[t] + a2b[t];
  }
}

// =====================================================================
// K5: mid term block. grid = 4 (one block per m).
// ci_mid[b,m,e,j] = leaf_out[b, m*4 + j/6, j%6, e]
// =====================================================================
__global__ __launch_bounds__(256) void mid_block_kernel(
    const float* __restrict__ leaf_out,
    const float* __restrict__ W, const float* __restrict__ bb,
    const float* __restrict__ gg, const float* __restrict__ bt,
    const float* __restrict__ gW, const float* __restrict__ gb,
    const float* __restrict__ a1W, const float* __restrict__ a1b,
    const float* __restrict__ a2w, const float* __restrict__ a2b,
    const float* __restrict__ drug_out,
    float* __restrict__ mid_out, float* __restrict__ mid_aux)
{
  const int m = blockIdx.x;
  const int tid = threadIdx.x;
  __shared__ float zs[6][2048];
  __shared__ float red[256], red2[256];
  __shared__ float mh[6], rsh[6];
  __shared__ float bnm[2048];
  __shared__ float gate[32];

  for (int idx = tid; idx < 6*2048; idx += 256) {
    int h = idx >> 11, be = idx & 2047, b = be >> 6, e = be & 63;
    const float* wp = W + (m*6 + h) * 24;
    float s = bb[m*6 + h];
#pragma unroll
    for (int j = 0; j < 24; ++j) {
      int tt = m*4 + (j/6), hh = j % 6;
      s += leaf_out[((size_t)(b*16 + tt) * 6 + hh) * 64 + e] * wp[j];
    }
    zs[h][be] = tanhf(s);
  }
  __syncthreads();
  for (int h = 0; h < 6; ++h) {
    float s = 0.f, s2 = 0.f;
    for (int i = tid; i < 2048; i += 256) { float v = zs[h][i]; s += v; s2 += v*v; }
    red[tid] = s; red2[tid] = s2; __syncthreads();
    for (int st = 128; st > 0; st >>= 1) {
      if (tid < st) { red[tid] += red[tid+st]; red2[tid] += red2[tid+st]; }
      __syncthreads();
    }
    if (tid == 0) {
      float mm = red[0] * (1.f/2048.f);
      float var = red2[0] * (1.f/2048.f) - mm*mm;
      mh[h] = mm; rsh[h] = rsqrtf(var + EPSV);
    }
    __syncthreads();
  }
  for (int be = tid; be < 2048; be += 256) {
    float s = 0.f;
#pragma unroll
    for (int h = 0; h < 6; ++h)
      s += (zs[h][be] - mh[h]) * rsh[h] * gg[m*6+h] + bt[m*6+h];
    bnm[be] = s * (1.f/6.f);
  }
  __syncthreads();
  if (tid < 32) {
    int b = tid;
    float s = gb[m];
    for (int e = 0; e < 64; ++e) s += bnm[b*64+e] * gW[m*70 + e];
    for (int j = 0; j < 6; ++j)  s += drug_out[b*6+j] * gW[m*70 + 64 + j];
    gate[b] = 1.f / (1.f + expf(-s));
  }
  __syncthreads();
  for (int idx = tid; idx < 6*2048; idx += 256) {
    int h = idx >> 11, be = idx & 2047, b = be >> 6, e = be & 63;
    float bn = (zs[h][be] - mh[h]) * rsh[h] * gg[m*6+h] + bt[m*6+h];
    mid_out[((size_t)(b*4 + m) * 6 + h) * 64 + e] = gate[b] * bn;
  }
  if (tid < 32) {
    int b = tid;
    float s = a1b[m];
    for (int e = 0; e < 64; ++e) s += gate[b] * bnm[b*64+e] * a1W[m*64+e];
    mid_aux[b*4 + m] = tanhf(s) * a2w[m] + a2b[m];
  }
}

// =====================================================================
// K6: root block. single workgroup.
// ci_root[b,e,c] = mid_out[b, c/6, c%6, e]
// =====================================================================
__global__ __launch_bounds__(256) void root_kernel(
    const float* __restrict__ mid_out,
    const float* rW, const float* rb, const float* rg, const float* rbt,
    const float* rgW, const float* rgb,
    const float* ra1W, const float* ra1b, const float* ra2w, const float* ra2b,
    const float* __restrict__ drug_out,
    float* __restrict__ root_mean, float* __restrict__ root_aux)
{
  const int tid = threadIdx.x;
  __shared__ float zs[6][2048];
  __shared__ float red[256], red2[256];
  __shared__ float mh[6], rsh[6];
  __shared__ float bnm[2048];
  __shared__ float gate[32];

  for (int idx = tid; idx < 6*2048; idx += 256) {
    int h = idx >> 11, be = idx & 2047, b = be >> 6, e = be & 63;
    float s = rb[h];
#pragma unroll
    for (int c = 0; c < 24; ++c)
      s += mid_out[((size_t)(b*4 + (c/6)) * 6 + (c%6)) * 64 + e] * rW[h*24 + c];
    zs[h][be] = tanhf(s);
  }
  __syncthreads();
  for (int h = 0; h < 6; ++h) {
    float s = 0.f, s2 = 0.f;
    for (int i = tid; i < 2048; i += 256) { float v = zs[h][i]; s += v; s2 += v*v; }
    red[tid] = s; red2[tid] = s2; __syncthreads();
    for (int st = 128; st > 0; st >>= 1) {
      if (tid < st) { red[tid] += red[tid+st]; red2[tid] += red2[tid+st]; }
      __syncthreads();
    }
    if (tid == 0) {
      float m = red[0] * (1.f/2048.f);
      float var = red2[0] * (1.f/2048.f) - m*m;
      mh[h] = m; rsh[h] = rsqrtf(var + EPSV);
    }
    __syncthreads();
  }
  for (int be = tid; be < 2048; be += 256) {
    float s = 0.f;
#pragma unroll
    for (int h = 0; h < 6; ++h)
      s += (zs[h][be] - mh[h]) * rsh[h] * rg[h] + rbt[h];
    bnm[be] = s * (1.f/6.f);
  }
  __syncthreads();
  if (tid < 32) {
    int b = tid;
    float s = rgb[0];
    for (int e = 0; e < 64; ++e) s += bnm[b*64+e] * rgW[e];
    for (int j = 0; j < 6; ++j)  s += drug_out[b*6+j] * rgW[64 + j];
    gate[b] = 1.f / (1.f + expf(-s));
  }
  __syncthreads();
  for (int idx = tid; idx < 2048; idx += 256)
    root_mean[idx] = gate[idx >> 6] * bnm[idx];
  if (tid < 32) {
    int b = tid;
    float s = ra1b[0];
    for (int e = 0; e < 64; ++e) s += gate[b] * bnm[b*64+e] * ra1W[e];
    root_aux[b] = tanhf(s) * ra2w[0] + ra2b[0];
  }
}

// =====================================================================
// K7: final head + output assembly. single workgroup.
// out (32,28) = [fout(6) | aux_final(1) | root_aux(1) | mid_aux(4) | leaf_aux(16)]
// =====================================================================
__global__ __launch_bounds__(256) void final_kernel(
    const float* __restrict__ root_mean, const float* __restrict__ drug_out,
    const float* fW, const float* fb, const float* fg, const float* fbt,
    const float* faW, const float* fab, const float* fow, const float* fob,
    const float* __restrict__ root_aux, const float* __restrict__ mid_aux,
    const float* __restrict__ leaf_aux, float* __restrict__ out)
{
  __shared__ float z[32*6];
  const int tid = threadIdx.x;
  for (int o = tid; o < 192; o += 256) {
    int b = o / 6, f = o % 6;
    float s = fb[f];
    for (int e = 0; e < 64; ++e) s += root_mean[b*64+e] * fW[f*70 + e];
    for (int j = 0; j < 6; ++j)  s += drug_out[b*6+j] * fW[f*70 + 64 + j];
    z[o] = tanhf(s);
  }
  __syncthreads();
  if (tid < 6) {
    int f = tid;
    float m = 0.f; for (int b = 0; b < 32; ++b) m += z[b*6+f]; m *= (1.f/32.f);
    float v = 0.f; for (int b = 0; b < 32; ++b) { float d = z[b*6+f]-m; v += d*d; } v *= (1.f/32.f);
    float rs = rsqrtf(v + EPSV);
    for (int b = 0; b < 32; ++b) z[b*6+f] = (z[b*6+f]-m)*rs*fg[f] + fbt[f];
  }
  __syncthreads();
  for (int o = tid; o < 192; o += 256) { int b = o/6, f = o%6; out[b*28 + f] = z[o]; }
  if (tid < 32) {
    int b = tid;
    float s = fab[0];
    for (int f = 0; f < 6; ++f) s += z[b*6+f] * faW[f];
    out[b*28 + 6] = tanhf(s) * fow[0] + fob[0];
    out[b*28 + 7] = root_aux[b];
  }
  for (int o = tid; o < 32*4; o += 256)  { int b = o/4,  t = o%4;  out[b*28 + 8 + t]  = mid_aux[b*4 + t]; }
  for (int o = tid; o < 32*16; o += 256) { int b = o/16, t = o%16; out[b*28 + 12 + t] = leaf_aux[b*16 + t]; }
}

// =====================================================================
extern "C" void kernel_launch(void* const* d_in, const int* in_sizes, int n_in,
                              void* d_out, int out_size, void* d_ws, size_t ws_size,
                              hipStream_t stream)
{
  (void)in_sizes; (void)n_in; (void)out_size; (void)ws_size;
  // setup_inputs dict order: x, table, then params in insertion order
  const float* x       = (const float*)d_in[0];
  const int*   table   = (const int*)  d_in[1];
  const float* rbf_W   = (const float*)d_in[2];
  const float* rbf_b   = (const float*)d_in[3];
  const float* emb     = (const float*)d_in[4];
  const float* dW0     = (const float*)d_in[5];
  const float* db0     = (const float*)d_in[6];
  const float* dg0     = (const float*)d_in[7];
  const float* dbt0    = (const float*)d_in[8];
  const float* dW1     = (const float*)d_in[9];
  const float* db1     = (const float*)d_in[10];
  const float* dg1     = (const float*)d_in[11];
  const float* dbt1    = (const float*)d_in[12];
  const float* dW2     = (const float*)d_in[13];
  const float* db2     = (const float*)d_in[14];
  const float* dg2     = (const float*)d_in[15];
  const float* dbt2    = (const float*)d_in[16];
  const float* leaf_dg_W = (const float*)d_in[17];
  const float* leaf_dg_b = (const float*)d_in[18];
  const float* leaf_W  = (const float*)d_in[19];
  const float* leaf_b  = (const float*)d_in[20];
  const float* leaf_g  = (const float*)d_in[21];
  const float* leaf_bt = (const float*)d_in[22];
  const float* leaf_gW = (const float*)d_in[23];
  const float* leaf_gb = (const float*)d_in[24];
  const float* leaf_a1W= (const float*)d_in[25];
  const float* leaf_a1b= (const float*)d_in[26];
  const float* leaf_a2w= (const float*)d_in[27];
  const float* leaf_a2b= (const float*)d_in[28];
  const float* mid_W   = (const float*)d_in[29];
  const float* mid_b   = (const float*)d_in[30];
  const float* mid_g   = (const float*)d_in[31];
  const float* mid_bt  = (const float*)d_in[32];
  const float* mid_gW  = (const float*)d_in[33];
  const float* mid_gb  = (const float*)d_in[34];
  const float* mid_a1W = (const float*)d_in[35];
  const float* mid_a1b = (const float*)d_in[36];
  const float* mid_a2w = (const float*)d_in[37];
  const float* mid_a2b = (const float*)d_in[38];
  const float* root_W  = (const float*)d_in[39];
  const float* root_b  = (const float*)d_in[40];
  const float* root_g  = (const float*)d_in[41];
  const float* root_bt = (const float*)d_in[42];
  const float* root_gW = (const float*)d_in[43];
  const float* root_gb = (const float*)d_in[44];
  const float* root_a1W= (const float*)d_in[45];
  const float* root_a1b= (const float*)d_in[46];
  const float* root_a2w= (const float*)d_in[47];
  const float* root_a2b= (const float*)d_in[48];
  const float* fW      = (const float*)d_in[49];
  const float* fb      = (const float*)d_in[50];
  const float* fg      = (const float*)d_in[51];
  const float* fbt     = (const float*)d_in[52];
  const float* faW     = (const float*)d_in[53];
  const float* fab     = (const float*)d_in[54];
  const float* fow     = (const float*)d_in[55];
  const float* fob     = (const float*)d_in[56];

  char* ws = (char*)d_ws;
  _Float16* rbfW_pack = (_Float16*)(ws + OFF_RBFP);
  _Float16* leafW_pack= (_Float16*)(ws + OFF_LEAFP);
  _Float16* gperm_h   = (_Float16*)(ws + OFF_GPERM);
  float* ci_leaf   = (float*)(ws + OFF_CILEAF);
  float* drug_out  = (float*)(ws + OFF_DRUG);
  float* leaf_out  = (float*)(ws + OFF_LEAFOUT);
  float* leaf_aux  = (float*)(ws + OFF_LEAFAUX);
  float* mid_out   = (float*)(ws + OFF_MIDOUT);
  float* mid_aux   = (float*)(ws + OFF_MIDAUX);
  float* root_mean = (float*)(ws + OFF_RMEAN);
  float* root_aux  = (float*)(ws + OFF_RAUX);

  pack_weights_kernel<<<512, 256, 0, stream>>>(rbf_W, leaf_dg_W, rbfW_pack, leafW_pack);
  drug_mlp_kernel<<<1, 256, 0, stream>>>(x, dW0, db0, dg0, dbt0,
                                         dW1, db1, dg1, dbt1,
                                         dW2, db2, dg2, dbt2, drug_out);
  rbf_wmma_kernel<<<dim3(16, 32), 256, 0, stream>>>(x, table, emb, rbf_b, rbfW_pack, gperm_h);
  leaf_wmma_kernel<<<dim3(16, 8), 256, 0, stream>>>(gperm_h, leafW_pack, ci_leaf);
  leaf_block_kernel<<<16, 256, 0, stream>>>(ci_leaf, leaf_dg_b, leaf_W, leaf_b, leaf_g, leaf_bt,
                                            leaf_gW, leaf_gb, leaf_a1W, leaf_a1b, leaf_a2w, leaf_a2b,
                                            drug_out, leaf_out, leaf_aux);
  mid_block_kernel<<<4, 256, 0, stream>>>(leaf_out, mid_W, mid_b, mid_g, mid_bt,
                                          mid_gW, mid_gb, mid_a1W, mid_a1b, mid_a2w, mid_a2b,
                                          drug_out, mid_out, mid_aux);
  root_kernel<<<1, 256, 0, stream>>>(mid_out, root_W, root_b, root_g, root_bt,
                                     root_gW, root_gb, root_a1W, root_a1b, root_a2w, root_a2b,
                                     drug_out, root_mean, root_aux);
  final_kernel<<<1, 256, 0, stream>>>(root_mean, drug_out, fW, fb, fg, fbt,
                                      faW, fab, fow, fob,
                                      root_aux, mid_aux, leaf_aux, (float*)d_out);
}
